// ManifoldConstrainedResidual_7971459301821
// MI455X (gfx1250) — matrix operand order repaired and checked by reading, not compiled, not run on hardware
//
#include <hip/hip_runtime.h>
#include <hip/hip_bf16.h>
#include <math.h>

#define B_ROWS 8192
#define S_DIM  4
#define D_DIM  1024
#define SD     4096
#define EPS_SK 1e-8f
#define EPS_LN 1e-5f
#define N_ITER 20

// Try the CDNA5 async Global->LDS path (ASYNCcnt-tracked). Set to 0 to fall
// back to plain loads if the assembler rejects the mnemonic.
#define USE_ASYNC_LDS 1

typedef __attribute__((ext_vector_type(16))) __bf16 v16bf;
typedef __attribute__((ext_vector_type(8)))  float  v8f;

__device__ __forceinline__ unsigned short f32_to_bf16(float f) {
  union { float f; unsigned int u; } c; c.f = f;
  unsigned int u = c.u;
  unsigned int r = u + 0x7FFFu + ((u >> 16) & 1u);   // round-to-nearest-even
  if ((u & 0x7FFFFFFFu) > 0x7F800000u) r = u | 0x00400000u; // quieten NaN
  return (unsigned short)(r >> 16);
}

// ---------------------------------------------------------------- K0: W_layer fp32 -> bf16
__global__ __launch_bounds__(256)
void k_convert_w(const float* __restrict__ W, unsigned short* __restrict__ Wb, int n) {
  int i = blockIdx.x * 256 + threadIdx.x;
  if (i < n) Wb[i] = f32_to_bf16(W[i]);
}

// ---------------------------------------------------------------- K1: per-row fused front end
// LN(4096) -> 24 dots (W_pre/W_post/W_res) -> sigmoids + sinkhorn -> x_agg -> LN(1024)*gamma+beta -> bf16
__global__ __launch_bounds__(256)
void k_row_fused(const float* __restrict__ x_streams,
                 const float* __restrict__ W_pre,  const float* __restrict__ b_pre,
                 const float* __restrict__ W_post, const float* __restrict__ b_post,
                 const float* __restrict__ W_res,  const float* __restrict__ b_res,
                 const float* __restrict__ alpha_pre, const float* __restrict__ alpha_post,
                 const float* __restrict__ alpha_res,
                 const float* __restrict__ gamma, const float* __restrict__ beta,
                 unsigned short* __restrict__ A_bf16,
                 float* __restrict__ h_post_out, float* __restrict__ h_res_out)
{
  __shared__ float xs[SD];          // raw row (needed again for x_agg)
  __shared__ float part[24 * 256];  // reduction scratch
  __shared__ float hpre[4];

  const int tid = threadIdx.x;
  const int b   = blockIdx.x;

  const float* xr = x_streams + (size_t)b * SD;

#if USE_ASYNC_LDS
  // GLOBAL_LOAD_ASYNC_TO_LDS_B128 (GVS mode): saddr = uniform row base (SGPR pair),
  // vaddr = per-lane 32-bit byte offset, vdst = LDS byte address.
  // 16384 bytes / (256 threads * 16B) = 4 chunks per thread.
#pragma unroll
  for (int q = 0; q < 4; ++q) {
    const unsigned byte_off = (unsigned)(tid * 16 + q * 4096);
    const unsigned lds_addr = (unsigned)(size_t)((char*)xs + byte_off);
    asm volatile("global_load_async_to_lds_b128 %0, %1, %2"
                 :: "v"(lds_addr), "v"(byte_off), "s"(xr)
                 : "memory");
  }
  asm volatile("s_wait_asynccnt 0x0" ::: "memory");
  __syncthreads();
#else
  for (int i = tid; i < SD; i += 256) xs[i] = xr[i];
  __syncthreads();
#endif

  // ---- stats for LN over 4096
  float s0 = 0.f, s1 = 0.f;
  for (int i = tid; i < SD; i += 256) { float v = xs[i]; s0 += v; s1 += v * v; }
  part[tid] = s0; part[256 + tid] = s1;
  __syncthreads();
  for (int off = 128; off > 0; off >>= 1) {
    if (tid < off) { part[tid] += part[tid + off]; part[256 + tid] += part[256 + tid + off]; }
    __syncthreads();
  }
  const float mu   = part[0]   * (1.0f / SD);
  const float var  = part[256] * (1.0f / SD) - mu * mu;
  const float rstd = rsqrtf(var + EPS_LN);
  __syncthreads();

  // ---- 24 dot products of x_norm with weight rows (L2-resident weights)
  float acc[24];
#pragma unroll
  for (int j = 0; j < 24; ++j) acc[j] = 0.f;
  for (int d = tid; d < SD; d += 256) {
    const float xn = (xs[d] - mu) * rstd;
#pragma unroll
    for (int j = 0; j < 4;  ++j) acc[j]     += xn * W_pre [j * SD + d];
#pragma unroll
    for (int j = 0; j < 4;  ++j) acc[4 + j] += xn * W_post[j * SD + d];
#pragma unroll
    for (int j = 0; j < 16; ++j) acc[8 + j] += xn * W_res [j * SD + d];
  }
#pragma unroll
  for (int j = 0; j < 24; ++j) part[j * 256 + tid] = acc[j];
  __syncthreads();
  for (int off = 128; off > 0; off >>= 1) {
    if (tid < off) {
#pragma unroll
      for (int j = 0; j < 24; ++j) part[j * 256 + tid] += part[j * 256 + tid + off];
    }
    __syncthreads();
  }

  // ---- heads + sinkhorn (tiny, one lane)
  if (tid == 0) {
    const float a_pre = alpha_pre[0], a_post = alpha_post[0], a_res = alpha_res[0];
#pragma unroll
    for (int s = 0; s < 4; ++s)
      hpre[s] = 1.0f / (1.0f + expf(-(a_pre * (part[s * 256] + b_pre[s]))));
#pragma unroll
    for (int s = 0; s < 4; ++s)
      h_post_out[b * 4 + s] = 2.0f / (1.0f + expf(-(a_post * (part[(4 + s) * 256] + b_post[s]))));
    float M[16];
#pragma unroll
    for (int j = 0; j < 16; ++j) M[j] = expf(a_res * (part[(8 + j) * 256] + b_res[j]));
    for (int it = 0; it < N_ITER; ++it) {
#pragma unroll
      for (int s = 0; s < 4; ++s) {
        float inv = 1.0f / (M[s*4] + M[s*4+1] + M[s*4+2] + M[s*4+3] + EPS_SK);
        M[s*4] *= inv; M[s*4+1] *= inv; M[s*4+2] *= inv; M[s*4+3] *= inv;
      }
#pragma unroll
      for (int t = 0; t < 4; ++t) {
        float inv = 1.0f / (M[t] + M[4+t] + M[8+t] + M[12+t] + EPS_SK);
        M[t] *= inv; M[4+t] *= inv; M[8+t] *= inv; M[12+t] *= inv;
      }
    }
#pragma unroll
    for (int j = 0; j < 16; ++j) h_res_out[b * 16 + j] = M[j];
  }
  __syncthreads();

  // ---- x_agg + LN(1024) affine -> bf16 activation row
  const float h0 = hpre[0], h1 = hpre[1], h2 = hpre[2], h3 = hpre[3];
  float aggv[4];
  s0 = 0.f; s1 = 0.f;
#pragma unroll
  for (int q = 0; q < 4; ++q) {
    const int d = tid + q * 256;
    const float a = h0 * xs[d] + h1 * xs[D_DIM + d] + h2 * xs[2 * D_DIM + d] + h3 * xs[3 * D_DIM + d];
    aggv[q] = a; s0 += a; s1 += a * a;
  }
  __syncthreads();
  part[tid] = s0; part[256 + tid] = s1;
  __syncthreads();
  for (int off = 128; off > 0; off >>= 1) {
    if (tid < off) { part[tid] += part[tid + off]; part[256 + tid] += part[256 + tid + off]; }
    __syncthreads();
  }
  const float mu2   = part[0]   * (1.0f / D_DIM);
  const float var2  = part[256] * (1.0f / D_DIM) - mu2 * mu2;
  const float rstd2 = rsqrtf(var2 + EPS_LN);
  unsigned short* arow = A_bf16 + (size_t)b * D_DIM;
#pragma unroll
  for (int q = 0; q < 4; ++q) {
    const int d = tid + q * 256;
    arow[d] = f32_to_bf16((aggv[q] - mu2) * rstd2 * gamma[d] + beta[d]);
  }
}

// ---------------------------------------------------------------- K2: Out = A(bf16) * W^T(bf16) + bias
// Each wave owns a 16x32 output tile (two 16x16 N-tiles sharing one A fragment).
// K-loop = 32 steps of 2x v_wmma_f32_16x16x32_bf16.
// A frag (16-bit A 16x32): lane m=lane&15, half=lane>>4; elems 0..15 contiguous at k0+half*16 in A row m.
// B frag (16-bit B 32x16): lane n=lane&15, half=lane>>4; elems 0..15 contiguous at k0+half*16 in W row n.
__global__ __launch_bounds__(256)
void k_gemm_wmma(const unsigned short* __restrict__ A,   // [B,1024] bf16
                 const unsigned short* __restrict__ Wb,  // [1024,1024] bf16, row = out feature
                 const float* __restrict__ bias,
                 float* __restrict__ Out)                // [B,1024] f32
{
  const int tid  = threadIdx.x;
  const int lane = tid & 31;
  const int wave = tid >> 5;
  const int cBlk = blockIdx.x & 7;    // 8 col blocks of 128
  const int rBlk = blockIdx.x >> 3;   // 256 row blocks of 32
  const int rb   = rBlk * 32 + (wave >> 2) * 16;
  const int cb   = cBlk * 128 + (wave & 3) * 32;
  const int m    = lane & 15;
  const int half = lane >> 4;

  const unsigned short* aRow  = A  + (size_t)(rb + m) * D_DIM;
  const unsigned short* bRow0 = Wb + (size_t)(cb + m) * D_DIM;        // n == lane&15
  const unsigned short* bRow1 = Wb + (size_t)(cb + 16 + m) * D_DIM;

  v8f acc0 = {}, acc1 = {};
  for (int k0 = 0; k0 < D_DIM; k0 += 32) {
    union { uint4 q[2]; v16bf v; } ua, ub0, ub1;
    ua.q[0] = *(const uint4*)(aRow + k0 + half * 16);
    ua.q[1] = *(const uint4*)(aRow + k0 + half * 16 + 8);
    const uint4* bp0 = (const uint4*)(bRow0 + k0 + half * 16);
    ub0.q[0] = bp0[0];
    ub0.q[1] = bp0[1];
    const uint4* bp1 = (const uint4*)(bRow1 + k0 + half * 16);
    ub1.q[0] = bp1[0];
    ub1.q[1] = bp1[1];
    acc0 = __builtin_amdgcn_wmma_f32_16x16x32_bf16(
        false, ua.v, false, ub0.v, (short)0, acc0, false, false);
    acc1 = __builtin_amdgcn_wmma_f32_16x16x32_bf16(
        false, ua.v, false, ub1.v, (short)0, acc1, false, false);
  }

  const float bv0 = bias[cb + m];
  const float bv1 = bias[cb + 16 + m];
#pragma unroll
  for (int v = 0; v < 8; ++v) {
    const int mm = v + 8 * half;                       // C/D layout: VGPR v -> M = v + 8*half
    float* orow = Out + (size_t)(rb + mm) * D_DIM;
    orow[cb + m]      = acc0[v] + bv0;
    orow[cb + 16 + m] = acc1[v] + bv1;
  }
}

// ---------------------------------------------------------------- K3: stream mix + outputs
__global__ __launch_bounds__(256)
void k_finalize(const float* __restrict__ x_streams,
                const float* __restrict__ OutLin,
                const float* __restrict__ h_post,
                const float* __restrict__ h_res,
                float* __restrict__ out_mean,      // [B,D]
                float* __restrict__ out_updated)   // [B,S,D]
{
  __shared__ float hr[16];
  __shared__ float hp[4];
  const int b = blockIdx.x, tid = threadIdx.x;
  if (tid < 16) hr[tid] = h_res[b * 16 + tid];
  if (tid < 4)  hp[tid] = h_post[b * 4 + tid];
  __syncthreads();
  const float* xr   = x_streams + (size_t)b * SD;
  const float* orow = OutLin    + (size_t)b * D_DIM;
  float* urow = out_updated + (size_t)b * SD;
  float* mrow = out_mean    + (size_t)b * D_DIM;
#pragma unroll
  for (int q = 0; q < 4; ++q) {
    const int d = tid + q * 256;
    const float x0 = xr[d], x1 = xr[D_DIM + d], x2 = xr[2 * D_DIM + d], x3 = xr[3 * D_DIM + d];
    const float o = orow[d];
    float msum = 0.f;
#pragma unroll
    for (int s = 0; s < 4; ++s) {
      const float u = hr[s*4+0]*x0 + hr[s*4+1]*x1 + hr[s*4+2]*x2 + hr[s*4+3]*x3 + o * hp[s];
      urow[s * D_DIM + d] = u;
      msum += u;
    }
    mrow[d] = msum * 0.25f;
  }
}

// ----------------------------------------------------------------
extern "C" void kernel_launch(void* const* d_in, const int* in_sizes, int n_in,
                              void* d_out, int out_size, void* d_ws, size_t ws_size,
                              hipStream_t stream) {
  const float* x_streams = (const float*)d_in[1];
  const float* W_pre     = (const float*)d_in[2];
  const float* b_pre     = (const float*)d_in[3];
  const float* W_post    = (const float*)d_in[4];
  const float* b_post    = (const float*)d_in[5];
  const float* W_res     = (const float*)d_in[6];
  const float* b_res     = (const float*)d_in[7];
  const float* alpha_pre  = (const float*)d_in[8];
  const float* alpha_post = (const float*)d_in[9];
  const float* alpha_res  = (const float*)d_in[10];
  const float* gamma   = (const float*)d_in[11];
  const float* beta    = (const float*)d_in[12];
  const float* W_layer = (const float*)d_in[13];
  const float* b_layer = (const float*)d_in[14];

  char* ws = (char*)d_ws;
  unsigned short* Wb = (unsigned short*)ws;  ws += (size_t)D_DIM * D_DIM * 2;   // 2 MB
  unsigned short* Ab = (unsigned short*)ws;  ws += (size_t)B_ROWS * D_DIM * 2;  // 16 MB
  float* OutLin = (float*)ws;                ws += (size_t)B_ROWS * D_DIM * 4;  // 32 MB
  float* HPost  = (float*)ws;                ws += (size_t)B_ROWS * 4 * 4;      // 128 KB
  float* HRes   = (float*)ws;                                                  // 512 KB

  k_convert_w<<<(D_DIM * D_DIM + 255) / 256, 256, 0, stream>>>(W_layer, Wb, D_DIM * D_DIM);

  k_row_fused<<<B_ROWS, 256, 0, stream>>>(x_streams, W_pre, b_pre, W_post, b_post,
                                          W_res, b_res, alpha_pre, alpha_post, alpha_res,
                                          gamma, beta, Ab, HPost, HRes);

  k_gemm_wmma<<<(B_ROWS / 32) * (D_DIM / 128), 256, 0, stream>>>(Ab, Wb, b_layer, OutLin);

  float* out_mean = (float*)d_out;
  float* out_upd  = out_mean + (size_t)B_ROWS * D_DIM;
  k_finalize<<<B_ROWS, 256, 0, stream>>>(x_streams, OutLin, HPost, HRes, out_mean, out_upd);
}
